// Face2Node_6528350290204
// MI455X (gfx1250) — matrix operand order repaired and checked by reading, not compiled, not run on hardware
//
#include <hip/hip_runtime.h>

typedef __attribute__((ext_vector_type(2)))  float  v2f;
typedef __attribute__((ext_vector_type(8)))  float  v8f;
typedef __attribute__((ext_vector_type(8)))  __bf16 v8bf;
typedef __attribute__((ext_vector_type(16))) __bf16 v16bf;

#define N_NODES 100000
#define N_FACES 200000
#define NTILES  (N_FACES / 16)
#define XP      136        // padded LDS K-pitch (elements): 272B = 68 dwords -> conflict-free row access
#define WAVES   4

// ---------- helpers ----------
__device__ __forceinline__ unsigned short bfh(float x) {
  return (unsigned short)(__float_as_uint(x) >> 16);   // truncating f32 -> bf16 (exact hi part)
}
__device__ __forceinline__ float bfF(unsigned short h) {
  return __uint_as_float(((unsigned)h) << 16);
}
__device__ __forceinline__ v8f splat8(float v) { v8f r = {v,v,v,v,v,v,v,v}; return r; }

// generic -> LDS byte offset (for async-to-LDS asm operands)
__device__ __forceinline__ unsigned ldsAddr(const void* p) {
  return (unsigned)(unsigned long long)(__attribute__((address_space(3))) const char*)p;
}

// A-operand fragment (16x32 bf16): lane holds 16 consecutive K
__device__ __forceinline__ v16bf ldA16(const unsigned short* p, int e) {
  v8bf a = *(const v8bf*)(p + e);
  v8bf b = *(const v8bf*)(p + e + 8);
  return __builtin_shufflevector(a, b, 0,1,2,3,4,5,6,7,8,9,10,11,12,13,14,15);
}
// B-operand fragment (32x16 bf16): lane holds 16 consecutive K
__device__ __forceinline__ v16bf ldB16(const unsigned short* p, int e) {
  v8bf a = *(const v8bf*)(p + e);
  v8bf b = *(const v8bf*)(p + e + 8);
  return __builtin_shufflevector(a, b, 0,1,2,3,4,5,6,7,8,9,10,11,12,13,14,15);
}

#define WMMA_BF16(A,B,C) __builtin_amdgcn_wmma_f32_16x16x32_bf16(false,(A),false,(B),(short)0,(C),false,false)
#define WMMA_F32(A,B,C)  __builtin_amdgcn_wmma_f32_16x16x4_f32(false,(A),false,(B),(short)0,(C),false,false)
#define WAVE_LDS_FENCE()   asm volatile("s_wait_dscnt 0x0" ::: "memory")
#define WAIT_ASYNC_ZERO()  asm volatile("s_wait_asynccnt 0x0" ::: "memory")

// async prefetch of one 16x128 f32 ff tile (8KB contiguous) into LDS raw buffer
__device__ __forceinline__ void async_ff_prefetch(const float* gtile, unsigned lds_base, int lane) {
  const unsigned long long g0 = (unsigned long long)(const char*)gtile + (unsigned)(lane * 16);
  const unsigned l0 = lds_base + (unsigned)(lane * 16);
  #pragma unroll
  for (int i = 0; i < 16; ++i) {
    asm volatile("global_load_async_to_lds_b128 %0, %1, off"
                 :: "v"(l0 + (unsigned)(i * 512)), "v"(g0 + (unsigned long long)(i * 512))
                 : "memory");
  }
}

// ---------- kernel 1: zero scratch + build transposed bf16 hi/lo weights ----------
__global__ void prep_kernel(const float* __restrict__ W1, const float* __restrict__ W2,
                            float* __restrict__ sums, float* __restrict__ counts,
                            unsigned short* __restrict__ w1h, unsigned short* __restrict__ w1l,
                            unsigned short* __restrict__ w2h, unsigned short* __restrict__ w2l) {
  const int TOT = 300000 + 100000 + 128*128 + 32*128;
  for (int i = blockIdx.x * blockDim.x + threadIdx.x; i < TOT; i += gridDim.x * blockDim.x) {
    if (i < 300000) {
      sums[i] = 0.0f;
    } else if (i < 400000) {
      counts[i - 300000] = 0.0f;
    } else if (i < 400000 + 128*128) {
      int t = i - 400000; int n = t >> 7, k = t & 127;
      float w = W1[(size_t)(k + 6) * 128 + n];          // ff rows of W1, transposed [n][k]
      unsigned short h = bfh(w);
      w1h[n*128 + k] = h;
      w1l[n*128 + k] = bfh(w - bfF(h));
    } else {
      int t = i - (400000 + 128*128); int n = t >> 7, k = t & 127;
      float w = W2[(size_t)k * 32 + n];                 // W2 transposed [n][k]
      unsigned short h = bfh(w);
      w2h[n*128 + k] = h;
      w2l[n*128 + k] = bfh(w - bfF(h));
    }
  }
}

// ---------- kernel 2: node degrees ----------
__global__ void count_kernel(const int* __restrict__ faces, float* __restrict__ counts) {
  for (int i = blockIdx.x * blockDim.x + threadIdx.x; i < N_FACES * 3; i += gridDim.x * blockDim.x) {
    atomicAdd(&counts[faces[i]], 1.0f);
  }
}

// ---------- kernel 3: the MLP + scatter ----------
__global__ __launch_bounds__(32 * WAVES, 1)
void face_mlp_kernel(const float* __restrict__ pos, const float* __restrict__ ff,
                     const float* __restrict__ W1, const float* __restrict__ b1,
                     const float* __restrict__ b2, const int* __restrict__ faces,
                     const unsigned short* __restrict__ w1h, const unsigned short* __restrict__ w1l,
                     const unsigned short* __restrict__ w2h, const unsigned short* __restrict__ w2l,
                     float* __restrict__ sums, float* __restrict__ outff) {
  __shared__ unsigned short sW1h[128 * XP];   // W1ff^T hi  [n][k]
  __shared__ unsigned short sW1l[128 * XP];   // W1ff^T lo
  __shared__ unsigned short sW2h[32 * XP];    // W2^T hi    [n][k]
  __shared__ unsigned short sW2l[32 * XP];    // W2^T lo
  __shared__ float          sW1e[6 * 128];    // W1 edge rows, row-major [k][n]
  __shared__ float          sB1[128];
  __shared__ float          sB2[32];
  __shared__ float          sFFraw[WAVES][16 * 128];  // per-wave raw f32 ff tile (async-to-LDS dest)
  __shared__ unsigned short sXh[WAVES][16 * XP];      // per-wave x / h staging (hi)
  __shared__ unsigned short sXl[WAVES][16 * XP];      // per-wave x / h staging (lo)
  __shared__ int            sNode[WAVES][16 * 3];

  for (int i = threadIdx.x; i < 128 * 128; i += blockDim.x) {
    int n = i >> 7, k = i & 127;
    sW1h[n*XP + k] = w1h[i];
    sW1l[n*XP + k] = w1l[i];
  }
  for (int i = threadIdx.x; i < 32 * 128; i += blockDim.x) {
    int n = i >> 7, k = i & 127;
    sW2h[n*XP + k] = w2h[i];
    sW2l[n*XP + k] = w2l[i];
  }
  for (int i = threadIdx.x; i < 6 * 128; i += blockDim.x) sW1e[i] = W1[i];
  if (threadIdx.x < 128) sB1[threadIdx.x] = b1[threadIdx.x];
  if (threadIdx.x < 32)  sB2[threadIdx.x] = b2[threadIdx.x];
  __syncthreads();

  const int wv   = threadIdx.x >> 5;
  const int lane = threadIdx.x & 31;
  const int mrow = lane & 15;     // matrix row (face within tile) / B column
  const int hh   = lane >> 4;     // lane half
  unsigned short* xh = &sXh[wv][0];
  unsigned short* xl = &sXl[wv][0];
  const float*    rawf = &sFFraw[wv][0];
  const unsigned  rawLds = ldsAddr(rawf);
  const int       stride = gridDim.x * WAVES;

  int tile = blockIdx.x * WAVES + wv;
  if (tile < NTILES) async_ff_prefetch(ff + (size_t)tile * 16 * 128, rawLds, lane);

  for (; tile < NTILES; tile += stride) {
    const int face0 = tile * 16;

    // ---- wait for async ff tile, convert raw f32 -> bf16 hi/lo in LDS ----
    WAIT_ASYNC_ZERO();
    #pragma unroll 4
    for (int rr = 0; rr < 16; ++rr) {
      const float4 v = *(const float4*)(rawf + rr * 128 + lane * 4);
      unsigned short h0 = bfh(v.x), h1 = bfh(v.y), h2 = bfh(v.z), h3 = bfh(v.w);
      unsigned short l0 = bfh(v.x - bfF(h0)), l1 = bfh(v.y - bfF(h1));
      unsigned short l2 = bfh(v.z - bfF(h2)), l3 = bfh(v.w - bfF(h3));
      unsigned* hp = (unsigned*)(xh + rr*XP + lane*4);
      unsigned* lp = (unsigned*)(xl + rr*XP + lane*4);
      hp[0] = (unsigned)h0 | ((unsigned)h1 << 16);
      hp[1] = (unsigned)h2 | ((unsigned)h3 << 16);
      lp[0] = (unsigned)l0 | ((unsigned)l1 << 16);
      lp[1] = (unsigned)l2 | ((unsigned)l3 << 16);
    }
    WAVE_LDS_FENCE();   // raw buffer fully consumed; bf16 staging visible

    // ---- software pipeline: kick off next tile's ff fetch behind this tile's compute ----
    {
      const int nxt = tile + stride;
      if (nxt < NTILES) async_ff_prefetch(ff + (size_t)nxt * 16 * 128, rawLds, lane);
    }

    // ---- face node indices & positions for this lane's row ----
    const int fbase = (face0 + mrow) * 3;
    const int i0 = faces[fbase + 0], i1 = faces[fbase + 1], i2 = faces[fbase + 2];
    if (hh == 0) {
      sNode[wv][mrow*3 + 0] = i0;
      sNode[wv][mrow*3 + 1] = i1;
      sNode[wv][mrow*3 + 2] = i2;
    }
    const float p0x = pos[i0*3+0], p0y = pos[i0*3+1], p0z = pos[i0*3+2];
    const float p1x = pos[i1*3+0], p1y = pos[i1*3+1], p1z = pos[i1*3+2];
    const float p2x = pos[i2*3+0], p2y = pos[i2*3+1], p2z = pos[i2*3+2];

    // ---- layer-1 base: b1 + ff @ W1ff  (bf16 hi/lo split, 3 products) ----
    v8f acc[8];
    #pragma unroll
    for (int j = 0; j < 8; ++j) acc[j] = splat8(sB1[j*16 + mrow]);

    #pragma unroll
    for (int kk = 0; kk < 4; ++kk) {
      const int ka = kk*32 + hh*16;
      const v16bf Ah = ldA16(xh, mrow*XP + ka);
      const v16bf Al = ldA16(xl, mrow*XP + ka);
      const int kb = kk*32 + hh*16;
      #pragma unroll
      for (int j = 0; j < 8; ++j) {
        const int nb = (j*16 + mrow)*XP + kb;
        const v16bf Bh = ldB16(sW1h, nb);
        const v16bf Bl = ldB16(sW1l, nb);
        acc[j] = WMMA_BF16(Ah, Bh, acc[j]);
        acc[j] = WMMA_BF16(Ah, Bl, acc[j]);
        acc[j] = WMMA_BF16(Al, Bh, acc[j]);
      }
    }

    v8f ffacc0 = splat8(0.0f);
    v8f ffacc1 = splat8(0.0f);

    // ---- per-corner: edge term (exact f32 WMMA), relu, layer-2 ----
    #pragma unroll 1
    for (int c = 0; c < 3; ++c) {
      float ax, ay, az, bx, by, bz;
      if (c == 0)      { ax = p1x-p0x; ay = p1y-p0y; az = p1z-p0z; bx = p2x-p0x; by = p2y-p0y; bz = p2z-p0z; }
      else if (c == 1) { ax = p2x-p1x; ay = p2y-p1y; az = p2z-p1z; bx = p0x-p1x; by = p0y-p1y; bz = p0z-p1z; }
      else             { ax = p0x-p2x; ay = p0y-p2y; az = p0z-p2z; bx = p1x-p2x; by = p1y-p2y; bz = p1z-p2z; }

      v2f ea0, ea1;
      if (hh) { ea0[0] = az; ea0[1] = bx; ea1[0] = 0.0f; ea1[1] = 0.0f; }
      else    { ea0[0] = ax; ea0[1] = ay; ea1[0] = by;   ea1[1] = bz;   }

      // pre-activation = acc + edge @ W1[0:6,:]; relu; restage as bf16 hi/lo (A layout for layer-2)
      #pragma unroll
      for (int j = 0; j < 8; ++j) {
        const int n = j*16 + mrow;
        v2f eb0, eb1;
        if (hh) { eb0[0] = sW1e[2*128 + n]; eb0[1] = sW1e[3*128 + n]; eb1[0] = 0.0f; eb1[1] = 0.0f; }
        else    { eb0[0] = sW1e[0*128 + n]; eb0[1] = sW1e[1*128 + n]; eb1[0] = sW1e[4*128 + n]; eb1[1] = sW1e[5*128 + n]; }
        v8f pre = WMMA_F32(ea0, eb0, acc[j]);
        pre     = WMMA_F32(ea1, eb1, pre);
        #pragma unroll
        for (int r = 0; r < 8; ++r) {
          const float hv = fmaxf(pre[r], 0.0f);
          const unsigned short hs = bfh(hv);
          const unsigned short ls = bfh(hv - bfF(hs));
          const int addr = (r + hh*8)*XP + n;   // h[row][feature]
          xh[addr] = hs;
          xl[addr] = ls;
        }
      }
      WAVE_LDS_FENCE();

      // layer-2: g = b2 + h @ W2 (bf16 hi/lo split)
      v8f g0 = splat8(sB2[mrow]);
      v8f g1 = splat8(sB2[16 + mrow]);
      #pragma unroll
      for (int kk = 0; kk < 4; ++kk) {
        const int ka = kk*32 + hh*16;
        const v16bf Ah = ldA16(xh, mrow*XP + ka);
        const v16bf Al = ldA16(xl, mrow*XP + ka);
        const int kb = kk*32 + hh*16;
        {
          const int nb = mrow*XP + kb;
          const v16bf Bh = ldB16(sW2h, nb);
          const v16bf Bl = ldB16(sW2l, nb);
          g0 = WMMA_BF16(Ah, Bh, g0);
          g0 = WMMA_BF16(Ah, Bl, g0);
          g0 = WMMA_BF16(Al, Bh, g0);
        }
        {
          const int nb = (16 + mrow)*XP + kb;
          const v16bf Bh = ldB16(sW2h, nb);
          const v16bf Bl = ldB16(sW2l, nb);
          g1 = WMMA_BF16(Ah, Bh, g1);
          g1 = WMMA_BF16(Ah, Bl, g1);
          g1 = WMMA_BF16(Al, Bh, g1);
        }
      }

      // scatter cols 0..2 to node sums
      if (mrow < 3) {
        #pragma unroll
        for (int r = 0; r < 8; ++r) {
          const int row  = r + hh*8;
          const int node = sNode[wv][row*3 + c];
          atomicAdd(&sums[node*3 + mrow], g0[r]);
        }
      }
      ffacc0 += g0;
      ffacc1 += g1;
    }

    // ---- face_features_out = mean over corners of cols 3..31 ----
    const float third = 1.0f / 3.0f;
    #pragma unroll
    for (int r = 0; r < 8; ++r) {
      const int row = r + hh*8;
      const size_t fo = (size_t)(face0 + row) * 29;
      if (mrow >= 3) outff[fo + (mrow - 3)] = ffacc0[r] * third;   // channels 0..12
      outff[fo + (13 + mrow)] = ffacc1[r] * third;                 // channels 13..28
    }
  }
}

// ---------- kernel 4: finalize node outputs ----------
__global__ void finalize_kernel(const float* __restrict__ pos, const float* __restrict__ sums,
                                const float* __restrict__ counts, float* __restrict__ out) {
  for (int i = blockIdx.x * blockDim.x + threadIdx.x; i < N_NODES; i += gridDim.x * blockDim.x) {
    const float inv = 1.0f / fmaxf(counts[i], 1.0f);
    #pragma unroll
    for (int d = 0; d < 3; ++d) {
      const float delta = sums[i*3 + d] * inv;
      out[i*3 + d]            = delta;                 // delta_pos
      out[300000 + i*3 + d]   = pos[i*3 + d] + delta;  // new_pos
    }
  }
}

extern "C" void kernel_launch(void* const* d_in, const int* in_sizes, int n_in,
                              void* d_out, int out_size, void* d_ws, size_t ws_size,
                              hipStream_t stream) {
  const float* pos   = (const float*)d_in[0];
  const float* ffeat = (const float*)d_in[1];
  const float* W1    = (const float*)d_in[2];
  const float* b1    = (const float*)d_in[3];
  const float* W2    = (const float*)d_in[4];
  const float* b2    = (const float*)d_in[5];
  const int*   faces = (const int*)d_in[6];
  float* out = (float*)d_out;

  // ws layout: sums[300000] f32 | counts[100000] f32 | W1T hi/lo, W2T hi/lo (bf16 as u16)
  float* sums   = (float*)d_ws;
  float* counts = sums + 300000;
  unsigned short* wbase = (unsigned short*)((char*)d_ws + 400000 * sizeof(float));
  unsigned short* w1h = wbase;
  unsigned short* w1l = wbase + 128*128;
  unsigned short* w2h = wbase + 2*128*128;
  unsigned short* w2l = wbase + 2*128*128 + 32*128;

  prep_kernel<<<1024, 256, 0, stream>>>(W1, W2, sums, counts, w1h, w1l, w2h, w2l);
  count_kernel<<<1024, 256, 0, stream>>>(faces, counts);
  face_mlp_kernel<<<1024, 32 * WAVES, 0, stream>>>(pos, ffeat, W1, b1, b2, faces,
                                                   w1h, w1l, w2h, w2l,
                                                   sums, out + 600000);
  finalize_kernel<<<512, 256, 0, stream>>>(pos, sums, counts, out);
}
